// SSMLayer_43319040147947
// MI455X (gfx1250) — matrix operand order repaired and checked by reading, not compiled, not run on hardware
//
#include <hip/hip_runtime.h>

// ---------------------------------------------------------------------------
// S4D SSM layer for MI455X (gfx1250, wave32, WMMA + TDM).
//   y[b,d,l] = Re( C @ scan_w( Bhat @ x ) )
// FFT conv replaced by exact linear recurrence (K is geometric).
// GEMM B-operands staged block-wide in LDS via tensor_load_to_lds (TDM).
// ---------------------------------------------------------------------------

typedef __attribute__((ext_vector_type(16))) _Float16 v16h;
typedef __attribute__((ext_vector_type(8)))  _Float16 v8h;
typedef __attribute__((ext_vector_type(8)))  float    v8f;
typedef __attribute__((ext_vector_type(4)))  unsigned int v4u;
typedef __attribute__((ext_vector_type(8)))  int      v8i;
typedef __attribute__((ext_vector_type(4)))  int      v4i;

#define TOT    512     // NUM_STATES * STATE_BLOCKS
#define C_IN   256
#define C_OUT  256
#define BATCH  8
#define SEQ    2048
#define NCHUNK 16
#define CLEN   128     // SEQ / NCHUNK

// ---- workspace layout (bytes) ----
#define WS_W2      ((size_t)0)                      // 512*4 f32 (wr, wi, wr^128, wi^128)
#define WS_BHAT    ((size_t)8192)                   // 512*256 f16
#define WS_CH      (WS_BHAT + (size_t)512*256*2)    // 256*512 f16
#define WS_XT      (WS_CH   + (size_t)256*512*2)    // 8*2048*256 f16  (xT[b][l][c])
#define WS_UT      (WS_XT   + (size_t)BATCH*SEQ*C_IN*2)   // 8*2048*512 f32 (uT[b][l][n])
#define WS_CONV    (WS_UT   + (size_t)BATCH*SEQ*TOT*4)    // 8*2048*512 f16 (convT[b][l][n])
#define WS_E       (WS_CONV + (size_t)BATCH*SEQ*TOT*2)    // 8*16*512 float2 (chunk states)

__device__ __forceinline__ v16h load_frag16(const _Float16* __restrict__ p) {
  v8h lo = *(const v8h*)(p);
  v8h hi = *(const v8h*)(p + 8);
  return __builtin_shufflevector(lo, hi, 0,1,2,3,4,5,6,7,8,9,10,11,12,13,14,15);
}
__device__ __forceinline__ v16h load_frag16_2(const _Float16* __restrict__ p0,
                                              const _Float16* __restrict__ p1) {
  v8h lo = *(const v8h*)(p0);
  v8h hi = *(const v8h*)(p1);
  return __builtin_shufflevector(lo, hi, 0,1,2,3,4,5,6,7,8,9,10,11,12,13,14,15);
}

#define WMMA(A, B, Cacc) \
  __builtin_amdgcn_wmma_f32_16x16x32_f16(false, (A), false, (B), (short)0, (Cacc), false, false)

// ---------------------------------------------------------------------------
// TDM: async 2D tile load (rows x rowlen f16, row stride = stride_elems)
// from global to LDS.  D# built per CDNA5 ISA ch.8 (2D tensor, groups 2/3=0).
// ---------------------------------------------------------------------------
__device__ __forceinline__ void tdm_load_2d_f16(unsigned long long gaddr,
                                                unsigned lds_off,
                                                unsigned rows,
                                                unsigned rowlen,
                                                unsigned stride_elems) {
  v4u g0;
  g0[0] = 1u;                                            // count=1, user mode
  g0[1] = lds_off;                                       // lds_addr (bytes)
  g0[2] = (unsigned)(gaddr & 0xffffffffu);               // global_addr[31:0]
  g0[3] = (unsigned)((gaddr >> 32) & 0x1ffffffu)         // global_addr[56:32]
        | (2u << 30);                                    // type = 2 ("image")
  v8i g1;
  g1[0] = (int)(1u << 16);                               // data_size=1 (2 bytes)
  g1[1] = (int)((rowlen & 0xffffu) << 16);               // tensor_dim0[15:0]
  g1[2] = (int)(((rowlen >> 16) & 0xffffu)               // tensor_dim0[31:16]
        | ((rows & 0xffffu) << 16));                     // tensor_dim1[15:0]
  g1[3] = (int)(((rows >> 16) & 0xffffu)                 // tensor_dim1[31:16]
        | ((rowlen & 0xffffu) << 16));                   // tile_dim0
  g1[4] = (int)(rows & 0xffffu);                         // tile_dim1 (tile_dim2=0)
  g1[5] = (int)stride_elems;                             // tensor_dim0_stride[31:0]
  g1[6] = 0;                                             // strides hi = 0
  g1[7] = 0;
  v4i z = {0, 0, 0, 0};
#if __clang_major__ >= 23
  v8i z8 = {0, 0, 0, 0, 0, 0, 0, 0};
  __builtin_amdgcn_tensor_load_to_lds(g0, g1, z, z, z8, 0);
#else
  __builtin_amdgcn_tensor_load_to_lds(g0, g1, z, z, 0);
#endif
}

// ---------------------------------------------------------------------------
// 0a. Per-state params: w = exp(dt*(-softplus(A0)) + i*dt*A1), plus w^128.
// ---------------------------------------------------------------------------
__global__ __launch_bounds__(256) void params_kernel(const float* __restrict__ A,
                                                     const float* __restrict__ log_dt,
                                                     float* __restrict__ w4) {
  int n = blockIdx.x * 256 + threadIdx.x;
  if (n >= TOT) return;
  float a0 = A[2 * n + 0];
  float a1 = A[2 * n + 1];
  float Ar = -log1pf(expf(a0));          // -softplus
  float dt = expf(log_dt[n]);
  float mag = expf(dt * Ar);
  float wr = mag * cosf(dt * a1);
  float wi = mag * sinf(dt * a1);
  float mr = wr, mi = wi;                // w^128 by 7 squarings
  #pragma unroll
  for (int i = 0; i < 7; ++i) {
    float nr = mr * mr - mi * mi;
    float ni = 2.0f * mr * mi;
    mr = nr; mi = ni;
  }
  w4[4 * n + 0] = wr; w4[4 * n + 1] = wi;
  w4[4 * n + 2] = mr; w4[4 * n + 3] = mi;
}

// ---------------------------------------------------------------------------
// 0b. Bhat[n][c] = f16( B[n][c] * dt[n] )      0c. Ch = f16(C)
// ---------------------------------------------------------------------------
__global__ __launch_bounds__(256) void bhat_kernel(const float* __restrict__ B,
                                                   const float* __restrict__ log_dt,
                                                   _Float16* __restrict__ Bhat) {
  int idx = blockIdx.x * 256 + threadIdx.x;        // 512*256
  int n = idx >> 8;
  float dt = expf(log_dt[n]);
  Bhat[idx] = (_Float16)(B[idx] * dt);
}
__global__ __launch_bounds__(256) void castc_kernel(const float* __restrict__ C,
                                                    _Float16* __restrict__ Ch) {
  int idx = blockIdx.x * 256 + threadIdx.x;        // 256*512
  Ch[idx] = (_Float16)C[idx];
}

// ---------------------------------------------------------------------------
// 0d. xT[b][l][c] = f16( x[b][c][l] )  — LDS-tiled 32x32 transpose
// ---------------------------------------------------------------------------
__global__ __launch_bounds__(256) void transpose_kernel(const float* __restrict__ x,
                                                        _Float16* __restrict__ xT) {
  __shared__ float tile[32][33];
  int bid = blockIdx.x;                 // b * (C_IN/32) * (SEQ/32) = 8*8*64
  int b  = bid / (8 * 64);
  int r  = bid % (8 * 64);
  int ct = r / 64, lt = r % 64;
  int c0 = ct * 32, l0 = lt * 32;
  int tx = threadIdx.x & 31;
  int ty = threadIdx.x >> 5;
  #pragma unroll
  for (int p = 0; p < 4; ++p) {
    int c = ty + p * 8;
    tile[c][tx] = x[((size_t)b * C_IN + (c0 + c)) * SEQ + l0 + tx];
  }
  __syncthreads();
  #pragma unroll
  for (int p = 0; p < 4; ++p) {
    int l = ty + p * 8;
    xT[((size_t)b * SEQ + (l0 + l)) * C_IN + c0 + tx] = (_Float16)tile[tx][l];
  }
}

// ---------------------------------------------------------------------------
// 1. GEMM: uT[b][l][n] = sum_c Bhat[n][c] * xT[b][l][c]
//    Block (512 thr = 16 waves) covers all 512 n for one (b, 64-wide l tile).
//    The shared 64x256 f16 xT tile is TDM-staged in LDS once per block.
// ---------------------------------------------------------------------------
__global__ __launch_bounds__(512) void gemm_u_kernel(const _Float16* __restrict__ Bhat,
                                                     const _Float16* __restrict__ xT,
                                                     float* __restrict__ uT) {
  __shared__ _Float16 btile[64 * C_IN];          // 32 KB
  const int bid = blockIdx.x;                    // 8 * 32 = 256
  const int b  = bid >> 5;
  const int lt = bid & 31;
  const int l0 = lt * 64;
  const int wv   = threadIdx.x >> 5;             // 0..15
  const int lane = threadIdx.x & 31;
  const int m0 = wv * 32;
  const int half = lane >> 4, lan = lane & 15;

  if (wv == 0) {
    unsigned long long ga =
        (unsigned long long)(const void*)(xT + ((size_t)b * SEQ + l0) * C_IN);
    tdm_load_2d_f16(ga, (unsigned)(size_t)(void*)btile, 64, C_IN, C_IN);
    __builtin_amdgcn_s_wait_tensorcnt(0);
  }
  __syncthreads();

  v8f acc0 = {}, acc1 = {}, acc2 = {}, acc3 = {};
  v8f acc4 = {}, acc5 = {}, acc6 = {}, acc7 = {};
  const _Float16* arow0 = Bhat + (size_t)(m0 + lan) * C_IN;
  const _Float16* arow1 = Bhat + (size_t)(m0 + 16 + lan) * C_IN;
  const _Float16* t0 = &btile[(size_t)( 0 + lan) * C_IN + half * 16];
  const _Float16* t1 = &btile[(size_t)(16 + lan) * C_IN + half * 16];
  const _Float16* t2 = &btile[(size_t)(32 + lan) * C_IN + half * 16];
  const _Float16* t3 = &btile[(size_t)(48 + lan) * C_IN + half * 16];

  #pragma unroll
  for (int kb = 0; kb < C_IN / 32; ++kb) {
    const int kn = kb * 32;
    v16h a0 = load_frag16(arow0 + kn + half * 16);
    v16h a1 = load_frag16(arow1 + kn + half * 16);
    v16h f0 = load_frag16(t0 + kn);
    v16h f1 = load_frag16(t1 + kn);
    v16h f2 = load_frag16(t2 + kn);
    v16h f3 = load_frag16(t3 + kn);
    acc0 = WMMA(a0, f0, acc0);
    acc1 = WMMA(a0, f1, acc1);
    acc2 = WMMA(a0, f2, acc2);
    acc3 = WMMA(a0, f3, acc3);
    acc4 = WMMA(a1, f0, acc4);
    acc5 = WMMA(a1, f1, acc5);
    acc6 = WMMA(a1, f2, acc6);
    acc7 = WMMA(a1, f3, acc7);
  }

  float* ub = uT + (size_t)b * SEQ * TOT + m0 + half * 8;   // 32B aligned
  *(v8f*)(ub + (size_t)(l0 +  0 + lan) * TOT)      = acc0;
  *(v8f*)(ub + (size_t)(l0 + 16 + lan) * TOT)      = acc1;
  *(v8f*)(ub + (size_t)(l0 + 32 + lan) * TOT)      = acc2;
  *(v8f*)(ub + (size_t)(l0 + 48 + lan) * TOT)      = acc3;
  *(v8f*)(ub + (size_t)(l0 +  0 + lan) * TOT + 16) = acc4;
  *(v8f*)(ub + (size_t)(l0 + 16 + lan) * TOT + 16) = acc5;
  *(v8f*)(ub + (size_t)(l0 + 32 + lan) * TOT + 16) = acc6;
  *(v8f*)(ub + (size_t)(l0 + 48 + lan) * TOT + 16) = acc7;
}

// ---------------------------------------------------------------------------
// 2. Chunked complex scan (coalesced): s[l] = w*s[l-1] + u[l].
// ---------------------------------------------------------------------------
__global__ __launch_bounds__(256) void scan_local_kernel(const float* __restrict__ w4,
                                                         const float* __restrict__ uT,
                                                         float2* __restrict__ E) {
  int tid = blockIdx.x * 256 + threadIdx.x;   // 8*16*512 = 65536
  int n = tid & 511;
  int c = (tid >> 9) & 15;
  int b = tid >> 13;
  float wr = w4[4 * n + 0], wi = w4[4 * n + 1];
  const float* up = uT + ((size_t)b * SEQ + c * CLEN) * TOT + n;
  float sr = 0.f, si = 0.f;
  for (int j = 0; j < CLEN; ++j) {
    float u   = up[(size_t)j * TOT];
    float nsr = fmaf(wr, sr, fmaf(-wi, si, u));
    float nsi = fmaf(wr, si, wi * sr);
    sr = nsr; si = nsi;
  }
  E[tid] = make_float2(sr, si);
}

__global__ __launch_bounds__(256) void scan_carry_kernel(const float* __restrict__ w4,
                                                         float2* __restrict__ E) {
  int tid = blockIdx.x * 256 + threadIdx.x;   // 4096
  int n = tid & 511;
  int b = tid >> 9;
  float mr = w4[4 * n + 2], mi = w4[4 * n + 3];   // w^128
  float cr = 0.f, ci = 0.f;
  for (int c = 0; c < NCHUNK; ++c) {
    float2* slot = &E[((size_t)b * NCHUNK + c) * TOT + n];
    float2 e = *slot;
    *slot = make_float2(cr, ci);                 // carry-in for chunk c
    float nr = fmaf(mr, cr, fmaf(-mi, ci, e.x));
    float ni = fmaf(mr, ci, fmaf(mi, cr, e.y));
    cr = nr; ci = ni;
  }
}

__global__ __launch_bounds__(256) void scan_emit_kernel(const float* __restrict__ w4,
                                                        const float* __restrict__ uT,
                                                        const float2* __restrict__ E,
                                                        _Float16* __restrict__ convT) {
  int tid = blockIdx.x * 256 + threadIdx.x;   // 65536
  int n = tid & 511;
  int c = (tid >> 9) & 15;
  int b = tid >> 13;
  float wr = w4[4 * n + 0], wi = w4[4 * n + 1];
  float2 carry = E[tid];
  float sr = carry.x, si = carry.y;
  const float* up = uT + ((size_t)b * SEQ + c * CLEN) * TOT + n;
  _Float16* cp = convT + ((size_t)b * SEQ + c * CLEN) * TOT + n;
  for (int j = 0; j < CLEN; ++j) {
    float u   = up[(size_t)j * TOT];
    float nsr = fmaf(wr, sr, fmaf(-wi, si, u));
    float nsi = fmaf(wr, si, wi * sr);
    sr = nsr; si = nsi;
    cp[(size_t)j * TOT] = (_Float16)sr;
  }
}

// ---------------------------------------------------------------------------
// 3. GEMM: y[b][d][l] = sum_n Ch[d][n] * convT[b][l][n]
//    Block (256 thr = 8 waves) covers all 256 d for one (b, 64-wide l tile).
//    The shared 64x512 f16 convT tile is TDM-staged in LDS once per block.
// ---------------------------------------------------------------------------
__device__ __forceinline__ void store_y(float* __restrict__ ybase, int l, v8f a) {
  #pragma unroll
  for (int r = 0; r < 8; ++r) ybase[(size_t)r * SEQ + l] = a[r];
}

__global__ __launch_bounds__(256) void gemm_y_kernel(const _Float16* __restrict__ Ch,
                                                     const _Float16* __restrict__ convT,
                                                     float* __restrict__ y) {
  __shared__ _Float16 ctile[64 * TOT];           // 64 KB
  const int bid = blockIdx.x;                    // 8 * 32 = 256
  const int b  = bid >> 5;
  const int lt = bid & 31;
  const int l0 = lt * 64;
  const int wv   = threadIdx.x >> 5;             // 0..7
  const int lane = threadIdx.x & 31;
  const int m0 = wv * 32;
  const int half = lane >> 4, lan = lane & 15;

  if (wv == 0) {
    unsigned long long ga =
        (unsigned long long)(const void*)(convT + ((size_t)b * SEQ + l0) * TOT);
    tdm_load_2d_f16(ga, (unsigned)(size_t)(void*)ctile, 64, TOT, TOT);
    __builtin_amdgcn_s_wait_tensorcnt(0);
  }
  __syncthreads();

  v8f acc0 = {}, acc1 = {}, acc2 = {}, acc3 = {};
  v8f acc4 = {}, acc5 = {}, acc6 = {}, acc7 = {};
  const _Float16* arow0 = Ch + (size_t)(m0 + lan) * TOT;
  const _Float16* arow1 = Ch + (size_t)(m0 + 16 + lan) * TOT;
  const _Float16* t0 = &ctile[(size_t)( 0 + lan) * TOT + half * 16];
  const _Float16* t1 = &ctile[(size_t)(16 + lan) * TOT + half * 16];
  const _Float16* t2 = &ctile[(size_t)(32 + lan) * TOT + half * 16];
  const _Float16* t3 = &ctile[(size_t)(48 + lan) * TOT + half * 16];

  #pragma unroll
  for (int kb = 0; kb < TOT / 32; ++kb) {
    const int kn = kb * 32;
    v16h a0 = load_frag16(arow0 + kn + half * 16);
    v16h a1 = load_frag16(arow1 + kn + half * 16);
    v16h f0 = load_frag16(t0 + kn);
    v16h f1 = load_frag16(t1 + kn);
    v16h f2 = load_frag16(t2 + kn);
    v16h f3 = load_frag16(t3 + kn);
    acc0 = WMMA(a0, f0, acc0);
    acc1 = WMMA(a0, f1, acc1);
    acc2 = WMMA(a0, f2, acc2);
    acc3 = WMMA(a0, f3, acc3);
    acc4 = WMMA(a1, f0, acc4);
    acc5 = WMMA(a1, f1, acc5);
    acc6 = WMMA(a1, f2, acc6);
    acc7 = WMMA(a1, f3, acc7);
  }

  float* yb0 = y + ((size_t)b * C_OUT + (m0 + half * 8)) * SEQ;
  float* yb1 = y + ((size_t)b * C_OUT + (m0 + 16 + half * 8)) * SEQ;
  store_y(yb0, l0 +  0 + lan, acc0);
  store_y(yb0, l0 + 16 + lan, acc1);
  store_y(yb0, l0 + 32 + lan, acc2);
  store_y(yb0, l0 + 48 + lan, acc3);
  store_y(yb1, l0 +  0 + lan, acc4);
  store_y(yb1, l0 + 16 + lan, acc5);
  store_y(yb1, l0 + 32 + lan, acc6);
  store_y(yb1, l0 + 48 + lan, acc7);
}

// ---------------------------------------------------------------------------
extern "C" void kernel_launch(void* const* d_in, const int* in_sizes, int n_in,
                              void* d_out, int out_size, void* d_ws, size_t ws_size,
                              hipStream_t stream) {
  const float* x      = (const float*)d_in[0];   // (8, 256, 2048)
  const float* A      = (const float*)d_in[1];   // (512, 2)
  const float* B      = (const float*)d_in[2];   // (512, 256)
  const float* C      = (const float*)d_in[3];   // (256, 512)
  const float* log_dt = (const float*)d_in[4];   // (512,)
  float* y = (float*)d_out;

  char* ws = (char*)d_ws;
  float*    w4    = (float*)   (ws + WS_W2);
  _Float16* Bhat  = (_Float16*)(ws + WS_BHAT);
  _Float16* Ch    = (_Float16*)(ws + WS_CH);
  _Float16* xT    = (_Float16*)(ws + WS_XT);
  float*    uT    = (float*)   (ws + WS_UT);
  _Float16* convT = (_Float16*)(ws + WS_CONV);
  float2*   E     = (float2*)  (ws + WS_E);

  params_kernel    <<<   2, 256, 0, stream>>>(A, log_dt, w4);
  bhat_kernel      <<< 512, 256, 0, stream>>>(B, log_dt, Bhat);
  castc_kernel     <<< 512, 256, 0, stream>>>(C, Ch);
  transpose_kernel <<<4096, 256, 0, stream>>>(x, xT);
  gemm_u_kernel    <<< 256, 512, 0, stream>>>(Bhat, xT, uT);
  scan_local_kernel<<< 256, 256, 0, stream>>>(w4, uT, E);
  scan_carry_kernel<<<  16, 256, 0, stream>>>(w4, E);
  scan_emit_kernel <<< 256, 256, 0, stream>>>(w4, uT, E, convT);
  gemm_y_kernel    <<< 256, 256, 0, stream>>>(Ch, convT, y);
}
